// DiscreteAttnTRBlock_25520695673112
// MI455X (gfx1250) — compile-verified, matrix operand order checked by reading, not run on hardware
//
#include <hip/hip_runtime.h>
#include <hip/hip_bf16.h>

typedef __bf16 bf16_t;
typedef __attribute__((ext_vector_type(8)))  __bf16 v8bf;
typedef __attribute__((ext_vector_type(16))) __bf16 v16bf;
typedef __attribute__((ext_vector_type(8)))  float  v8f;

#define CIN   128      // C
#define CH2   256      // C*H
#define KNB   27
#define EPS   1e-5f
#define MT    16       // m-tiles processed per wave in the GEMM (B amortization)

// ---------------------------------------------------------------------------
// Convert x (fp32) -> bf16, grid-stride
__global__ void k_x2bf(const float* __restrict__ x, bf16_t* __restrict__ xbf, int total) {
    for (int i = blockIdx.x * blockDim.x + threadIdx.x; i < total; i += gridDim.x * blockDim.x)
        xbf[i] = (bf16_t)x[i];
}

// Prep: transpose+convert weights to bf16 (n-major so WMMA B frags are contiguous),
// compute ksum[m][k] = sum_c cb[m,k,c], zero the atomic stat accumulators.
__global__ void k_prep(const float* __restrict__ Wv, const float* __restrict__ Wout,
                       const float* __restrict__ cb,
                       bf16_t* __restrict__ WvT, bf16_t* __restrict__ WoutT,
                       float* __restrict__ ksum, float* __restrict__ zero_region, int zero_count) {
    int tid = blockIdx.x * blockDim.x + threadIdx.x;
    int stride = gridDim.x * blockDim.x;
    // WvT[n*128 + k] = Wv[k*256 + n]   (Wv is [128][256])
    for (int i = tid; i < CIN * CH2; i += stride) {
        int n = i / CIN, k = i % CIN;
        WvT[(size_t)n * CIN + k] = (bf16_t)Wv[(size_t)k * CH2 + n];
    }
    // WoutT[n*256 + k] = Wout[k*128 + n]   (Wout is [256][128])
    for (int i = tid; i < CIN * CH2; i += stride) {
        int n = i / CH2, k = i % CH2;
        WoutT[(size_t)n * CH2 + k] = (bf16_t)Wout[(size_t)k * CIN + n];
    }
    // ksum [2][27]
    for (int i = tid; i < 2 * KNB; i += stride) {
        float s = 0.f;
        const float* row = cb + (size_t)i * CH2;
        for (int c = 0; c < CH2; ++c) s += row[c];
        ksum[i] = s;
    }
    for (int i = tid; i < zero_count; i += stride) zero_region[i] = 0.f;
}

// ---------------------------------------------------------------------------
__device__ __forceinline__ v16bf pack_frag(const bf16_t* p) {
    v8bf lo = *(const v8bf*)(p);        // K = base + 0..7   (per half-wave)
    v8bf hi = *(const v8bf*)(p + 16);   // K = base + 16..23 (per half-wave)
    return __builtin_shufflevector(lo, hi, 0,1,2,3,4,5,6,7,8,9,10,11,12,13,14,15);
}

// bf16 WMMA GEMM: Cmat[N x NC] = A[N x K] * Bt[NC x K]^T, fp32 out,
// plus per-column sum / sumsq accumulated with global atomics.
// Block = one 16-col n-strip; B fragments register-resident for the wave's
// lifetime.  A fragments are double-buffered across the MT m-tiles so the
// loads for tile t+1 are in flight while tile t's WMMAs execute (partial
// loadcnt waits instead of loadcnt==0).  NC is a template constant so the 8
// epilogue stores use one base address + immediate offsets.
template <int KSTEPS, int NC>   // KSTEPS = K/32
__global__ void k_gemm_bn(const bf16_t* __restrict__ A, const bf16_t* __restrict__ Bt,
                          float* __restrict__ Cmat,
                          float* __restrict__ colsum, float* __restrict__ colssq) {
    constexpr int K = KSTEPS * 32;
    int lane = threadIdx.x & 31;
    int wv   = threadIdx.x >> 5;
    int tn   = blockIdx.x;                 // n tile (strip) for this block
    int half = lane >> 4;                  // K sub-block select (ISA 16-bit layout)
    int r    = lane & 15;

    // --- B fragments: loaded once, live in registers across all m-tiles ---
    const bf16_t* brow = Bt + (size_t)(tn * 16 + r) * K + half * 8;
    v16bf bfrag[KSTEPS];
#pragma unroll
    for (int s = 0; s < KSTEPS; ++s) bfrag[s] = pack_frag(brow + s * 32);

    int col = tn * 16 + r;
    int tm0 = (blockIdx.y * (blockDim.x >> 5) + wv) * MT;

    const bf16_t* arow = A + (size_t)(tm0 * 16 + r) * K + half * 8;
    const size_t astr = (size_t)16 * K;            // one m-tile of A per lane row

    v16bf bufA[KSTEPS], bufB[KSTEPS];

    auto load_tile = [&](const bf16_t* p, v16bf* dst) {
#pragma unroll
        for (int s = 0; s < KSTEPS; ++s) dst[s] = pack_frag(p + s * 32);
    };
    auto compute_store = [&](const v16bf* av, int t) {
        v8f c = {0.f, 0.f, 0.f, 0.f, 0.f, 0.f, 0.f, 0.f};
#pragma unroll
        for (int s = 0; s < KSTEPS; ++s)
            c = __builtin_amdgcn_wmma_f32_16x16x32_bf16(false, av[s], false, bfrag[s],
                                                        (short)0, c, false, false);
        // D layout: VGPR j -> row base + half*8 + j, col tn*16 + r.
        int rowbase = (tm0 + t) * 16 + half * 8;
        float* crow = Cmat + (size_t)rowbase * NC + col;
        float s1 = 0.f, s2 = 0.f;
#pragma unroll
        for (int j = 0; j < 8; ++j) {
            float vj = c[j];
            crow[j * NC] = vj;                     // immediate-offset stores
            s1 += vj;
            s2 += vj * vj;
        }
        // pair the two half-rows of each column -> one atomic per column per tile
        s1 += __shfl_xor(s1, 16, 32);
        s2 += __shfl_xor(s2, 16, 32);
        if (lane < 16) {
            atomicAdd(&colsum[col], s1);
            atomicAdd(&colssq[col], s2);
        }
    };

    load_tile(arow, bufA);
    for (int t = 0; t < MT; t += 2) {
        if (t + 2 < MT) {
            // WGP-scope prefetch (scope 0: fill all cache levels), 2 tiles ahead
            const bf16_t* pf = arow + 2 * astr;
            asm volatile("global_prefetch_b8 %0, off" :: "v"(pf));
        }
        load_tile(arow + astr, bufB);              // tile t+1 loads in flight
        compute_store(bufA, t);                    // consume tile t
        if (t + 2 < MT) load_tile(arow + 2 * astr, bufA);   // tile t+2 in flight
        compute_store(bufB, t + 1);                // consume tile t+1
        arow += 2 * astr;
    }
}

// Finalize BN: scale = g*rsqrt(var+eps), shift = b - mean*scale
__global__ void k_bnfin(const float* __restrict__ sum, const float* __restrict__ ssq,
                        const float* __restrict__ g, const float* __restrict__ b,
                        float* __restrict__ scale, float* __restrict__ shift,
                        int cols, float invN) {
    int c = blockIdx.x * blockDim.x + threadIdx.x;
    if (c >= cols) return;
    float mu  = sum[c] * invN;
    float var = ssq[c] * invN - mu * mu;
    float sc  = g[c] * rsqrtf(var + EPS);
    scale[c] = sc;
    shift[c] = b[c] - mu * sc;
}

// Apply BN + ReLU in place on y [N x cols]
__global__ void k_bnapply(float* __restrict__ y, const float* __restrict__ scale,
                          const float* __restrict__ shift, int cols, long total) {
    for (long i = (long)blockIdx.x * blockDim.x + threadIdx.x; i < total;
         i += (long)gridDim.x * blockDim.x) {
        int c = (int)(i % cols);
        y[i] = fmaxf(fmaf(y[i], scale[c], shift[c]), 0.f);
    }
}

// ---------------------------------------------------------------------------
// q kernel: wave-per-point.  q[p] = sum_k dot(x[nbr[p,k]], Wq[k]).
// Wq (27x128) staged in LDS.  Block-level stat reduction -> 2 atomics/block.
__global__ void k_q(const float* __restrict__ x, const int* __restrict__ nbr,
                    const float* __restrict__ Wq, float* __restrict__ qraw,
                    float* __restrict__ qsum, float* __restrict__ qssq) {
    __shared__ float sw[KNB * CIN];
    __shared__ float bsum, bssq;
    for (int i = threadIdx.x; i < KNB * CIN; i += blockDim.x) sw[i] = Wq[i];
    if (threadIdx.x == 0) { bsum = 0.f; bssq = 0.f; }
    __syncthreads();

    int lane = threadIdx.x & 31;
    int wv   = threadIdx.x >> 5;
    int p    = blockIdx.x * (blockDim.x >> 5) + wv;
    const int* nrow = nbr + (size_t)p * KNB;

    float acc = 0.f;
    for (int k = 0; k < KNB; ++k) {
        int idx = nrow[k];
        float4 xv = *(const float4*)(x + (size_t)idx * CIN + lane * 4);
        float4 wq = *(const float4*)(sw + k * CIN + lane * 4);
        acc += xv.x * wq.x + xv.y * wq.y + xv.z * wq.z + xv.w * wq.w;
    }
#pragma unroll
    for (int off = 16; off > 0; off >>= 1) acc += __shfl_xor(acc, off, 32);
    if (lane == 0) {
        qraw[p] = acc;
        atomicAdd(&bsum, acc);
        atomicAdd(&bssq, acc * acc);
    }
    __syncthreads();
    if (threadIdx.x == 0) {
        atomicAdd(qsum, bsum);
        atomicAdd(qssq, bssq);
    }
}

// choice kernel: thread-per-point, BN+ReLU on gathered q, 2-way softmax
__global__ void k_choice(const float* __restrict__ qraw, const int* __restrict__ nbr,
                         const float* __restrict__ ksum,
                         const float* __restrict__ qsc, const float* __restrict__ qsh,
                         float* __restrict__ choice) {
    __shared__ float sk[2 * KNB];
    if (threadIdx.x < 2 * KNB) sk[threadIdx.x] = ksum[threadIdx.x];
    __syncthreads();
    int p = blockIdx.x * blockDim.x + threadIdx.x;
    float sc = qsc[0], sh = qsh[0];
    const int* nrow = nbr + (size_t)p * KNB;
    float l0 = 0.f, l1 = 0.f;
    for (int k = 0; k < KNB; ++k) {
        float qb = fmaxf(fmaf(qraw[nrow[k]], sc, sh), 0.f);
        l0 += qb * sk[k];
        l1 += qb * sk[KNB + k];
    }
    float mx = fmaxf(l0, l1);
    float e0 = __expf(l0 - mx), e1 = __expf(l1 - mx);
    float inv = 1.f / (e0 + e1);
    choice[(size_t)p * 2]     = e0 * inv;
    choice[(size_t)p * 2 + 1] = e1 * inv;
}

// routed channelwise conv: wave-per-point, full 55KB codebook in LDS (dynamic),
// result written directly as bf16 (A-matrix of GEMM2).
__global__ void k_route(const float* __restrict__ v, const int* __restrict__ nbr,
                        const float* __restrict__ cb, const float* __restrict__ choice,
                        bf16_t* __restrict__ outbf) {
    extern __shared__ float scb[];                 // [2][27][256] = 13824 floats
    for (int i = threadIdx.x; i < 2 * KNB * CH2; i += blockDim.x) scb[i] = cb[i];
    __syncthreads();

    int lane = threadIdx.x & 31;
    int wv   = threadIdx.x >> 5;
    int p    = blockIdx.x * (blockDim.x >> 5) + wv;
    float ch0 = choice[(size_t)p * 2];
    float ch1 = choice[(size_t)p * 2 + 1];
    const int* nrow = nbr + (size_t)p * KNB;

    float acc[8] = {0.f, 0.f, 0.f, 0.f, 0.f, 0.f, 0.f, 0.f};
    for (int k = 0; k < KNB; ++k) {
        int idx = nrow[k];
        const float* vrow = v + (size_t)idx * CH2;
        const float* c0 = scb + k * CH2;
        const float* c1 = scb + (KNB + k) * CH2;
#pragma unroll
        for (int j = 0; j < 8; ++j) {
            int c = lane + 32 * j;                 // coalesced across the wave
            acc[j] = fmaf(vrow[c], fmaf(ch0, c0[c], ch1 * c1[c]), acc[j]);
        }
    }
    bf16_t* orow = outbf + (size_t)p * CH2;
#pragma unroll
    for (int j = 0; j < 8; ++j) orow[lane + 32 * j] = (bf16_t)acc[j];
}

// final: out = relu(bn(z)) + x
__global__ void k_final(const float* __restrict__ z, const float* __restrict__ x,
                        const float* __restrict__ scale, const float* __restrict__ shift,
                        float* __restrict__ out, long total) {
    for (long i = (long)blockIdx.x * blockDim.x + threadIdx.x; i < total;
         i += (long)gridDim.x * blockDim.x) {
        int c = (int)(i % CIN);
        out[i] = fmaxf(fmaf(z[i], scale[c], shift[c]), 0.f) + x[i];
    }
}

// ---------------------------------------------------------------------------
extern "C" void kernel_launch(void* const* d_in, const int* in_sizes, int n_in,
                              void* d_out, int out_size, void* d_ws, size_t ws_size,
                              hipStream_t stream) {
    const float* x    = (const float*)d_in[0];
    const int*   nbr  = (const int*)  d_in[1];
    const float* Wv   = (const float*)d_in[2];
    const float* gv   = (const float*)d_in[3];
    const float* bv   = (const float*)d_in[4];
    const float* Wq   = (const float*)d_in[5];
    const float* gq   = (const float*)d_in[6];
    const float* bq   = (const float*)d_in[7];
    const float* cb   = (const float*)d_in[8];
    const float* Wout = (const float*)d_in[9];
    const float* go   = (const float*)d_in[10];
    const float* bo   = (const float*)d_in[11];
    float* out = (float*)d_out;

    const int N = in_sizes[0] / CIN;               // 131072
    const float invN = 1.f / (float)N;

    // --- workspace carve-up ---
    char* w = (char*)d_ws;
    size_t off = 0;
    auto alloc = [&](size_t bytes) -> void* {
        off = (off + 255) & ~(size_t)255;
        void* p = w + off;
        off += bytes;
        return p;
    };
    bf16_t* xbf    = (bf16_t*)alloc((size_t)N * CIN * sizeof(bf16_t));
    bf16_t* WvT    = (bf16_t*)alloc((size_t)CH2 * CIN * sizeof(bf16_t));
    bf16_t* WoutT  = (bf16_t*)alloc((size_t)CIN * CH2 * sizeof(bf16_t));
    float*  y1     = (float*) alloc((size_t)N * CH2 * sizeof(float));  // v after bnapply (in place)
    float*  qraw   = (float*) alloc((size_t)N * sizeof(float));
    float*  choice = (float*) alloc((size_t)N * 2 * sizeof(float));
    bf16_t* outbf  = (bf16_t*)alloc((size_t)N * CH2 * sizeof(bf16_t));
    float*  z      = (float*) alloc((size_t)N * CIN * sizeof(float));
    float*  ksum   = (float*) alloc(2 * KNB * sizeof(float));
    // atomic accumulators (contiguous zero region): sum_v[256] ssq_v[256] qstat[2] sum_o[128] ssq_o[128]
    const int zero_count = CH2 + CH2 + 2 + CIN + CIN;
    float* zr     = (float*)alloc(zero_count * sizeof(float));
    float* sum_v  = zr;
    float* ssq_v  = zr + CH2;
    float* qsum   = zr + 2 * CH2;
    float* qssq   = zr + 2 * CH2 + 1;
    float* sum_o  = zr + 2 * CH2 + 2;
    float* ssq_o  = zr + 2 * CH2 + 2 + CIN;
    float* sc_v   = (float*)alloc(CH2 * sizeof(float));
    float* sh_v   = (float*)alloc(CH2 * sizeof(float));
    float* qscshf = (float*)alloc(2 * sizeof(float));   // [0]=scale, [1]=shift
    float* sc_o   = (float*)alloc(CIN * sizeof(float));
    float* sh_o   = (float*)alloc(CIN * sizeof(float));
    (void)ws_size; (void)n_in; (void)out_size;

    // --- phase 0: conversions / prep / zeroing ---
    k_x2bf<<<2048, 256, 0, stream>>>(x, xbf, N * CIN);
    k_prep<<<64, 256, 0, stream>>>(Wv, Wout, cb, WvT, WoutT, ksum, zr, zero_count);

    // --- phase 1: v-GEMM (bf16 WMMA, double-buffered A) + column stats ---
    {
        int mtiles = N / 16;                        // 8192
        dim3 grid(CH2 / 16, mtiles / (8 * MT));     // (16, 64)
        k_gemm_bn<CIN / 32, CH2><<<grid, 256, 0, stream>>>(xbf, WvT, y1, sum_v, ssq_v);
    }
    k_bnfin<<<1, 256, 0, stream>>>(sum_v, ssq_v, gv, bv, sc_v, sh_v, CH2, invN);
    k_bnapply<<<4096, 256, 0, stream>>>(y1, sc_v, sh_v, CH2, (long)N * CH2);  // y1 -> v

    // --- phase 2: q (gather dot), BN scalar, choice softmax ---
    k_q<<<N / 8, 256, 0, stream>>>(x, nbr, Wq, qraw, qsum, qssq);
    k_bnfin<<<1, 32, 0, stream>>>(qsum, qssq, gq, bq, &qscshf[0], &qscshf[1], 1, invN);
    k_choice<<<N / 256, 256, 0, stream>>>(qraw, nbr, ksum, &qscshf[0], &qscshf[1], choice);

    // --- phase 3: routed channelwise conv (codebook in LDS) -> bf16 ---
    k_route<<<N / 8, 256, 2 * KNB * CH2 * sizeof(float), stream>>>(y1, nbr, cb, choice, outbf);

    // --- phase 4: out-GEMM (bf16 WMMA, double-buffered A) + column stats ---
    {
        int mtiles = N / 16;                        // 8192
        dim3 grid(CIN / 16, mtiles / (8 * MT));     // (8, 64)
        k_gemm_bn<CH2 / 32, CIN><<<grid, 256, 0, stream>>>(outbf, WoutT, z, sum_o, ssq_o);
    }
    k_bnfin<<<1, 128, 0, stream>>>(sum_o, ssq_o, go, bo, sc_o, sh_o, CIN, invN);
    k_final<<<2048, 256, 0, stream>>>(z, x, sc_o, sh_o, out, (long)N * CIN);
}
